// Attention_12103217840348
// MI455X (gfx1250) — matrix operand order repaired and checked by reading, not compiled, not run on hardware
//
#include <hip/hip_runtime.h>
#include <hip/hip_bf16.h>

typedef __attribute__((ext_vector_type(16))) _Float16 v16h;
typedef __attribute__((ext_vector_type(8)))  _Float16 v8h;
typedef __attribute__((ext_vector_type(8)))  float    v8f;

#define BATCH 4
#define SEQ   2048
#define DIM   1024
#define HEADS 16
#define DHEAD 64
#define SCALE 0.125f   // 64^-0.5

// ---------------------------------------------------------------------------
// WMMA helpers (CDNA5 f16 -> f32, 16x16x32)
// ---------------------------------------------------------------------------
__device__ inline v8f wmma_f16(v16h a, v16h b, v8f c) {
    return __builtin_amdgcn_wmma_f32_16x16x32_f16(
        /*neg_a=*/false, a, /*neg_b=*/false, b,
        /*c_mod=*/(short)0, c, /*reuse_a=*/false, /*reuse_b=*/false);
}

// A-matrix fragment, 16x32 f16, row-major source with leading dim lda.
// Layout used here (contiguous k, not the ISA table order): lane L holds row M=L&15; kb=(L>>4)*16; elems j -> K=kb+j.
// One contiguous 32B load per lane.
__device__ inline v16h load_a_frag(const _Float16* A, int lda) {
    int lane = threadIdx.x & 31;
    int m  = lane & 15;
    int kb = (lane >> 4) * 16;
    const _Float16* p = A + (size_t)m * lda;
    return *(const v16h*)(p + kb);
}

// B-matrix fragment, 32x16 f16, from an N-major buffer Bt[n][k] (ldb = K stride).
// Layout used here (contiguous k, not the ISA table order): lane L holds col N=L&15; kb=(L>>4)*16; elems j -> K=kb+j.
// One contiguous 32B load per lane.
__device__ inline v16h load_b_frag(const _Float16* Bt, int ldb) {
    int lane = threadIdx.x & 31;
    int n  = lane & 15;
    int kb = (lane >> 4) * 16;
    return *(const v16h*)(Bt + (size_t)n * ldb + kb);
}

// ---------------------------------------------------------------------------
// Kernel 1: LayerNorm (f32 in, f16 out).  One block (256 thr) per row.
// ---------------------------------------------------------------------------
__global__ __launch_bounds__(256)
void k_layernorm(const float* __restrict__ x, const float* __restrict__ g,
                 const float* __restrict__ b, _Float16* __restrict__ out) {
    int row = blockIdx.x;
    const float* xr = x + (size_t)row * DIM;
    float v[4], lsum = 0.f, lsq = 0.f;
#pragma unroll
    for (int i = 0; i < 4; ++i) {
        v[i] = xr[i * 256 + threadIdx.x];
        lsum += v[i];
        lsq  += v[i] * v[i];
    }
#pragma unroll
    for (int off = 16; off > 0; off >>= 1) {
        lsum += __shfl_xor(lsum, off, 32);
        lsq  += __shfl_xor(lsq,  off, 32);
    }
    __shared__ float ssum[8], ssq[8];
    if ((threadIdx.x & 31) == 0) {
        ssum[threadIdx.x >> 5] = lsum;
        ssq [threadIdx.x >> 5] = lsq;
    }
    __syncthreads();
    float ts = 0.f, tq = 0.f;
#pragma unroll
    for (int i = 0; i < 8; ++i) { ts += ssum[i]; tq += ssq[i]; }
    float mu   = ts * (1.0f / DIM);
    float var  = tq * (1.0f / DIM) - mu * mu;
    float rstd = __frsqrt_rn(var + 1e-5f);
    _Float16* orow = out + (size_t)row * DIM;
#pragma unroll
    for (int i = 0; i < 4; ++i) {
        int c = i * 256 + threadIdx.x;
        orow[c] = (_Float16)((v[i] - mu) * rstd * g[c] + b[c]);
    }
}

// ---------------------------------------------------------------------------
// Kernel 2: W[rows][cols] f32 -> Wt[cols][rows] f16 (N-major for B frags)
// ---------------------------------------------------------------------------
__global__ __launch_bounds__(256)
void k_transpose_cast(const float* __restrict__ W, _Float16* __restrict__ Wt,
                      int rows, int cols) {
    long idx   = (long)blockIdx.x * 256 + threadIdx.x;
    long total = (long)rows * cols;
    if (idx < total) {
        int r = (int)(idx / cols), c = (int)(idx % cols);
        Wt[(size_t)c * rows + r] = (_Float16)W[idx];
    }
}

// ---------------------------------------------------------------------------
// Ping-pong double-buffered wave GEMM core: 64x32 wave tile.
// Fragment sets alternate roles every 32-K step (no register copies).
// The K-loop must stay rolled: a second compiler unroll doubles live fragment
// sets past the 256-VGPR window and forces accumulator shuffling.
// ---------------------------------------------------------------------------
__device__ inline void wave_gemm_64x32(const _Float16* __restrict__ Abase,
                                       const _Float16* __restrict__ Bbase,
                                       int K, v8f acc[4][2]) {
    v16h a0[4], b0[2], a1[4], b1[2];
#pragma unroll
    for (int i = 0; i < 4; ++i) a0[i] = load_a_frag(Abase + (size_t)(16 * i) * K, K);
#pragma unroll
    for (int j = 0; j < 2; ++j) b0[j] = load_b_frag(Bbase + (size_t)(16 * j) * K, K);

#pragma clang loop unroll(disable)
    for (int k = 0; k < K - 64; k += 64) {
#pragma unroll
        for (int i = 0; i < 4; ++i)
            a1[i] = load_a_frag(Abase + (size_t)(16 * i) * K + k + 32, K);
#pragma unroll
        for (int j = 0; j < 2; ++j)
            b1[j] = load_b_frag(Bbase + (size_t)(16 * j) * K + k + 32, K);
#pragma unroll
        for (int i = 0; i < 4; ++i)
#pragma unroll
            for (int j = 0; j < 2; ++j)
                acc[i][j] = wmma_f16(a0[i], b0[j], acc[i][j]);
#pragma unroll
        for (int i = 0; i < 4; ++i)
            a0[i] = load_a_frag(Abase + (size_t)(16 * i) * K + k + 64, K);
#pragma unroll
        for (int j = 0; j < 2; ++j)
            b0[j] = load_b_frag(Bbase + (size_t)(16 * j) * K + k + 64, K);
#pragma unroll
        for (int i = 0; i < 4; ++i)
#pragma unroll
            for (int j = 0; j < 2; ++j)
                acc[i][j] = wmma_f16(a1[i], b1[j], acc[i][j]);
    }
    // tail: tiles K-64 (in a0/b0) and K-32
#pragma unroll
    for (int i = 0; i < 4; ++i)
        a1[i] = load_a_frag(Abase + (size_t)(16 * i) * K + K - 32, K);
#pragma unroll
    for (int j = 0; j < 2; ++j)
        b1[j] = load_b_frag(Bbase + (size_t)(16 * j) * K + K - 32, K);
#pragma unroll
    for (int i = 0; i < 4; ++i)
#pragma unroll
        for (int j = 0; j < 2; ++j)
            acc[i][j] = wmma_f16(a0[i], b0[j], acc[i][j]);
#pragma unroll
    for (int i = 0; i < 4; ++i)
#pragma unroll
        for (int j = 0; j < 2; ++j)
            acc[i][j] = wmma_f16(a1[i], b1[j], acc[i][j]);
}

// ---------------------------------------------------------------------------
// Kernel 3: QKV GEMM  xn[8192][1024] @ Wt[3072][1024] (N-major).
// Block = 256 thr = 8 waves (2 M x 4 N); block tile 128x128; wave tile 64x32.
// Epilogue scatters to head-major Q[b][h][n][d], K[b][h][n][d], Vt[b][h][d][n];
// the Q/K/V selector is wave-uniform (32-col tile never crosses a 1024 edge).
// ---------------------------------------------------------------------------
__global__ __launch_bounds__(256)
void k_qkv_gemm(const _Float16* __restrict__ Xn, const _Float16* __restrict__ Wt,
                _Float16* __restrict__ Qd, _Float16* __restrict__ Kd,
                _Float16* __restrict__ Vt) {
    const int K = DIM;
    int wave = threadIdx.x >> 5;
    int wm = wave & 1, wn = wave >> 1;
    int m0 = blockIdx.y * 128 + wm * 64;
    int n0 = blockIdx.x * 128 + wn * 32;

    v8f acc[4][2] = {};
    wave_gemm_64x32(Xn + (size_t)m0 * K, Wt + (size_t)n0 * K, K, acc);

    int lane = threadIdx.x & 31;
    int col16 = lane & 15, hi = lane >> 4;
    int which = n0 >> 10;                 // wave-uniform: 0=Q 1=K 2=V
    int f0 = n0 & 1023;

    if (which < 2) {
        _Float16* __restrict__ dst = (which == 0) ? Qd : Kd;
#pragma unroll
        for (int i = 0; i < 4; ++i)
#pragma unroll
            for (int j = 0; j < 2; ++j)
#pragma unroll
                for (int r = 0; r < 8; ++r) {
                    int row = m0 + 16 * i + r + 8 * hi;
                    int f   = f0 + 16 * j + col16;
                    int h = f >> 6, d = f & 63;
                    int bidx = row >> 11, t = row & 2047;
                    size_t bh = (size_t)bidx * HEADS + h;
                    dst[(bh * SEQ + t) * DHEAD + d] = (_Float16)acc[i][j][r];
                }
    } else {
#pragma unroll
        for (int i = 0; i < 4; ++i)
#pragma unroll
            for (int j = 0; j < 2; ++j)
#pragma unroll
                for (int r = 0; r < 8; ++r) {
                    int row = m0 + 16 * i + r + 8 * hi;
                    int f   = f0 + 16 * j + col16;
                    int h = f >> 6, d = f & 63;
                    int bidx = row >> 11, t = row & 2047;
                    size_t bh = (size_t)bidx * HEADS + h;
                    Vt[(bh * DHEAD + d) * SEQ + t] = (_Float16)acc[i][j][r];
                }
    }
}

// ---------------------------------------------------------------------------
// Kernel 4: flash attention.  Grid (SEQ/16/8, HEADS, BATCH), 256 thr = 8 waves,
// each wave owns one 16-query tile and streams 32 keys/iteration.
// ---------------------------------------------------------------------------
__global__ __launch_bounds__(256)
void k_attention(const _Float16* __restrict__ Qd, const _Float16* __restrict__ Kd,
                 const _Float16* __restrict__ Vt, const float* __restrict__ alibi,
                 _Float16* __restrict__ Aout) {
    __shared__ __align__(32) _Float16 pbuf[8][16][32];   // per-wave P staging
    int wave = threadIdx.x >> 5, lane = threadIdx.x & 31;
    int qb = blockIdx.x * 8 + wave;                      // 16-query tile index
    int h = blockIdx.y, bidx = blockIdx.z;

    size_t bh = (size_t)bidx * HEADS + h;
    const _Float16* Qh = Qd + bh * SEQ * DHEAD;
    const _Float16* Kh = Kd + bh * SEQ * DHEAD;
    const _Float16* Vh = Vt + bh * DHEAD * SEQ;
    const float*    Ah = alibi + ((size_t)h * SEQ + qb * 16) * SEQ;

    v16h aq0 = load_a_frag(Qh + (size_t)qb * 16 * DHEAD,      DHEAD);
    v16h aq1 = load_a_frag(Qh + (size_t)qb * 16 * DHEAD + 32, DHEAD);

    v8f acc[4] = {};
    float mrow[8], lrow[8];
#pragma unroll
    for (int r = 0; r < 8; ++r) { mrow[r] = -1e30f; lrow[r] = 0.f; }

    int col = lane & 15, hi = lane >> 4;

#pragma clang loop unroll(disable)
    for (int jb = 0; jb < SEQ; jb += 32) {
        // prefetch next key/value tiles (gfx1250 global_prefetch_b8)
        if (jb + 32 < SEQ) {
            __builtin_prefetch((const char*)(Kh + (size_t)(jb + 32) * DHEAD) + lane * 128, 0, 3);
            __builtin_prefetch((const char*)(Vh + (size_t)lane * SEQ + jb + 32), 0, 3);
            __builtin_prefetch((const char*)(Vh + (size_t)(lane + 32) * SEQ + jb + 32), 0, 3);
        }
        // ---- scores: 16 queries x 32 keys, K(dim)=64 in two WMMA steps ----
        v8f s[2] = {};
#pragma unroll
        for (int nt = 0; nt < 2; ++nt) {
            v16h bk0 = load_b_frag(Kh + (size_t)(jb + nt * 16) * DHEAD,      DHEAD);
            v16h bk1 = load_b_frag(Kh + (size_t)(jb + nt * 16) * DHEAD + 32, DHEAD);
            s[nt] = wmma_f16(aq0, bk0, s[nt]);
            s[nt] = wmma_f16(aq1, bk1, s[nt]);
        }
        // ---- scale + alibi (C layout: row=r+8*hi, col=lane&15) ----
#pragma unroll
        for (int r = 0; r < 8; ++r) {
            int row = r + 8 * hi;
            s[0][r] = s[0][r] * SCALE + Ah[(size_t)row * SEQ + jb + col];
            s[1][r] = s[1][r] * SCALE + Ah[(size_t)row * SEQ + jb + 16 + col];
        }
        // ---- online softmax: row reductions across 16-lane groups ----
#pragma unroll
        for (int r = 0; r < 8; ++r) {
            float tm = fmaxf(s[0][r], s[1][r]);
#pragma unroll
            for (int off = 1; off < 16; off <<= 1)
                tm = fmaxf(tm, __shfl_xor(tm, off, 32));
            float mnew  = fmaxf(mrow[r], tm);
            float alpha = __expf(mrow[r] - mnew);
            mrow[r] = mnew;
            float p0 = __expf(s[0][r] - mnew);
            float p1 = __expf(s[1][r] - mnew);
            s[0][r] = p0; s[1][r] = p1;
            float ts = p0 + p1;
#pragma unroll
            for (int off = 1; off < 16; off <<= 1)
                ts += __shfl_xor(ts, off, 32);
            lrow[r] = lrow[r] * alpha + ts;
#pragma unroll
            for (int dt = 0; dt < 4; ++dt) acc[dt][r] *= alpha;
        }
        // ---- transpose P (C layout -> A layout) through LDS ----
#pragma unroll
        for (int r = 0; r < 8; ++r) {
            int row = r + 8 * hi;
            pbuf[wave][row][col]      = (_Float16)s[0][r];
            pbuf[wave][row][16 + col] = (_Float16)s[1][r];
        }
        asm volatile("s_wait_dscnt 0x0" ::: "memory");
        v16h pa = load_a_frag(&pbuf[wave][0][0], 32);
        // ---- out += P @ V  (B from Vt[d][n]: N-major in d, contiguous in n) --
#pragma unroll
        for (int dt = 0; dt < 4; ++dt) {
            v16h bv = load_b_frag(Vh + (size_t)(dt * 16) * SEQ + jb, SEQ);
            acc[dt] = wmma_f16(pa, bv, acc[dt]);
        }
    }
    // ---- epilogue: normalize by l, write b n (h d) layout ----
#pragma unroll
    for (int dt = 0; dt < 4; ++dt)
#pragma unroll
        for (int r = 0; r < 8; ++r) {
            int row = qb * 16 + r + 8 * hi;              // token within seq
            int d = dt * 16 + col;
            float o = acc[dt][r] / lrow[r];
            Aout[((size_t)bidx * SEQ + row) * DIM + h * DHEAD + d] = (_Float16)o;
        }
}

// ---------------------------------------------------------------------------
// Kernel 5: out projection  Aout[8192][1024] @ WoutT[1024][1024] + bias -> f32
// Same tiling as kernel 3, ping-pong double-buffered K-loop.
// ---------------------------------------------------------------------------
__global__ __launch_bounds__(256)
void k_out_gemm(const _Float16* __restrict__ Ain, const _Float16* __restrict__ Wt,
                const float* __restrict__ bias, float* __restrict__ Out) {
    const int K = DIM;
    int wave = threadIdx.x >> 5;
    int wm = wave & 1, wn = wave >> 1;
    int m0 = blockIdx.y * 128 + wm * 64;
    int n0 = blockIdx.x * 128 + wn * 32;

    v8f acc[4][2] = {};
    wave_gemm_64x32(Ain + (size_t)m0 * K, Wt + (size_t)n0 * K, K, acc);

    int lane = threadIdx.x & 31;
    int col16 = lane & 15, hi = lane >> 4;
#pragma unroll
    for (int i = 0; i < 4; ++i)
#pragma unroll
        for (int j = 0; j < 2; ++j)
#pragma unroll
            for (int r = 0; r < 8; ++r) {
                int row = m0 + 16 * i + r + 8 * hi;
                int col = n0 + 16 * j + col16;
                Out[(size_t)row * DIM + col] = acc[i][j][r] + bias[col];
            }
}

// ---------------------------------------------------------------------------
// Launcher
// ---------------------------------------------------------------------------
extern "C" void kernel_launch(void* const* d_in, const int* in_sizes, int n_in,
                              void* d_out, int out_size, void* d_ws, size_t ws_size,
                              hipStream_t stream) {
    const float* x      = (const float*)d_in[0];
    const float* alibi  = (const float*)d_in[1];
    const float* w_qkv  = (const float*)d_in[2];
    const float* w_out  = (const float*)d_in[3];
    const float* b_out  = (const float*)d_in[4];
    const float* ln_g   = (const float*)d_in[5];
    const float* ln_b   = (const float*)d_in[6];
    float* out = (float*)d_out;

    const size_t M = (size_t)BATCH * SEQ;            // 8192 tokens
    char* w = (char*)d_ws;
    _Float16* xn     = (_Float16*)w;  w += M * DIM * 2;                         // 16 MB
    _Float16* wqkv_t = (_Float16*)w;  w += (size_t)3 * DIM * DIM * 2;           //  6 MB
    _Float16* wout_t = (_Float16*)w;  w += (size_t)DIM * DIM * 2;               //  2 MB
    _Float16* Qd     = (_Float16*)w;  w += M * DIM * 2;                         // 16 MB
    _Float16* Kd     = (_Float16*)w;  w += M * DIM * 2;                         // 16 MB
    _Float16* Vt     = (_Float16*)w;  w += M * DIM * 2;                         // 16 MB
    _Float16* Aout   = (_Float16*)w;  w += M * DIM * 2;                         // 16 MB

    // 1. weight prep
    {
        long t1 = (long)DIM * 3 * DIM;
        k_transpose_cast<<<dim3((t1 + 255) / 256), dim3(256), 0, stream>>>(
            w_qkv, wqkv_t, DIM, 3 * DIM);
        long t2 = (long)DIM * DIM;
        k_transpose_cast<<<dim3((t2 + 255) / 256), dim3(256), 0, stream>>>(
            w_out, wout_t, DIM, DIM);
    }
    // 2. layernorm
    k_layernorm<<<dim3((unsigned)M), dim3(256), 0, stream>>>(x, ln_g, ln_b, xn);
    // 3. QKV GEMM: M=8192, N=3072, K=1024 -> grid (24, 64)
    k_qkv_gemm<<<dim3(3 * DIM / 128, (unsigned)(M / 128)), dim3(256), 0, stream>>>(
        xn, wqkv_t, Qd, Kd, Vt);
    // 4. attention: grid (SEQ/16/8, HEADS, BATCH)
    k_attention<<<dim3(SEQ / 16 / 8, HEADS, BATCH), dim3(256), 0, stream>>>(
        Qd, Kd, Vt, alibi, Aout);
    // 5. out projection: M=8192, N=1024, K=1024 -> grid (8, 64)
    k_out_gemm<<<dim3(DIM / 128, (unsigned)(M / 128)), dim3(256), 0, stream>>>(
        Aout, wout_t, b_out, out);
}
